// LDTR_68418829025668
// MI455X (gfx1250) — compile-verified, matrix operand order checked by reading, not run on hardware
//
#include <hip/hip_runtime.h>
#include <hip/hip_bf16.h>

// ---------------------------------------------------------------------------
// Mamba (selective scan) forward for 3 scales on gfx1250 (MI455X).
// GEMMs: v_wmma_f32_16x16x32_bf16, wave32. Each wave computes a 32x16 output
// tile (2 WMMAs per k-step sharing one LDS-staged, fragment-swizzled B tile).
// All GEMM A-operands are pre-converted to bf16 by the producer kernels, so
// the WMMA inner loop is pure b128 loads + ds_load_b128 + 2x wmma.
// K always a multiple of 32 (dt path zero-padded) -> no guards.
// ---------------------------------------------------------------------------

typedef __attribute__((ext_vector_type(16))) __bf16          v16bf;
typedef __attribute__((ext_vector_type(16))) unsigned short  v16us;
typedef __attribute__((ext_vector_type(8)))  unsigned short  v8us;
typedef __attribute__((ext_vector_type(8)))  float           v8f;

__device__ __forceinline__ unsigned short f2bf(float f) {
    unsigned int u = __float_as_uint(f);
    u += 0x7FFFu + ((u >> 16) & 1u);          // round-to-nearest-even
    return (unsigned short)(u >> 16);
}

__device__ __forceinline__ float silu_f(float x) {
    return x / (1.f + __expf(-x));
}

__device__ __forceinline__ float softplus_f(float x) {
    return (x > 20.f) ? x : log1pf(__expf(x));
}

// ------------------------------ weight cvt ---------------------------------
__global__ void f32_to_bf16_kernel(const float* __restrict__ src,
                                   unsigned short* __restrict__ dst,
                                   long long n) {
    long long g = (long long)blockIdx.x * 256 + threadIdx.x;
    if (g < n) dst[g] = f2bf(src[g]);
}

// zero-padded weight cvt: dst is (rows_dst, width), src is (rows_src, width)
__global__ void f32_to_bf16_pad_kernel(const float* __restrict__ src,
                                       unsigned short* __restrict__ dst,
                                       int rows_src, int width, long long n) {
    long long g = (long long)blockIdx.x * 256 + threadIdx.x;
    if (g >= n) return;
    int k = (int)(g / width);
    int c = (int)(g % width);
    dst[g] = (k < rows_src) ? f2bf(src[(size_t)k * width + c])
                            : (unsigned short)0;
}

// copy dt_r columns out of x_dbl into zero-padded (M, Kp) bf16 buffer
__global__ void pad_dtr_kernel(const float* __restrict__ xdbl,
                               unsigned short* __restrict__ dtr,
                               int rn, int r, int Kp, long long n) {
    long long g = (long long)blockIdx.x * 256 + threadIdx.x;
    if (g >= n) return;
    long long row = g / Kp;
    int k = (int)(g % Kp);
    dtr[g] = (k < r) ? f2bf(xdbl[row * (long long)rn + k]) : (unsigned short)0;
}

// ------------------------- rmsnorm + transpose -----------------------------
// x: (B, d, L) strided ; xn: (B*L, d) bf16 row-major, normalized * w
__global__ void __launch_bounds__(256)
rmsnorm_transpose_kernel(const float* __restrict__ x,
                         const float* __restrict__ w,
                         unsigned short* __restrict__ xn, int d, int L) {
    int l = blockIdx.x, b = blockIdx.y;
    __shared__ float sbuf[1024];
    __shared__ float sred[256];
    const float* xp = x + ((size_t)b * d) * L + l;
    float ss = 0.f;
    for (int c = threadIdx.x; c < d; c += 256) {
        float v = xp[(size_t)c * L];
        sbuf[c] = v;
        ss += v * v;
    }
    sred[threadIdx.x] = ss;
    __syncthreads();
    for (int s = 128; s > 0; s >>= 1) {
        if (threadIdx.x < s) sred[threadIdx.x] += sred[threadIdx.x + s];
        __syncthreads();
    }
    float rinv = rsqrtf(sred[0] / (float)d + 1e-5f);
    unsigned short* op = xn + ((size_t)b * L + l) * d;
    for (int c = threadIdx.x; c < d; c += 256)
        op[c] = f2bf(sbuf[c] * rinv * w[c]);
}

// ------------------------------ WMMA GEMM ----------------------------------
// C[M,N] = A[M,K](bf16, lda) * B[K,N](bf16, ldb)  (+bias, +softplus)
// Requirements: K % 32 == 0, N % 16 == 0, M % 128 == 0, lda % 8 == 0.
// block = 128 threads = 4 waves; each wave one 32x16 tile; block = 128x16.
__global__ void __launch_bounds__(128)
wmma_gemm_kernel(const unsigned short* __restrict__ A, int lda,
                 const unsigned short* __restrict__ Bw, int ldb,
                 float* __restrict__ Cc, int ldc,
                 int K,
                 const float* __restrict__ bias, int act) {
    // B tile (32x16 bf16) staged in "fragment order": lane's 16 elems contiguous
    __shared__ __align__(32) unsigned short sB[512];

    const int lane  = threadIdx.x & 31;
    const int wave  = threadIdx.x >> 5;
    const int tileN = blockIdx.x * 16;
    const int tileM = blockIdx.y * 128 + wave * 32;
    const int nloc  = lane & 15;
    const int lhi   = lane >> 4;               // 0 or 1
    const int khalf = lhi << 3;                // 0 or 8 (ISA lane layout)

    const unsigned short* a0 = A + (size_t)(tileM + nloc) * lda;       // rows 0..15
    const unsigned short* a1 = A + (size_t)(tileM + 16 + nloc) * lda;  // rows 16..31

    // staging coords for this thread: one uint2 (4 bf16) per k-step
    const int skk = threadIdx.x >> 2;          // 0..31  (k within tile)
    const int snn = (threadIdx.x & 3) << 2;    // 0,4,8,12 (n within tile)
    const int kmid = (skk >> 3) & 1, khi2 = skk >> 4, klo = skk & 7;
    const int sbase = khi2 * 8 + klo;          // element slot within fragment

    v8f acc0 = {0.f, 0.f, 0.f, 0.f, 0.f, 0.f, 0.f, 0.f};
    v8f acc1 = {0.f, 0.f, 0.f, 0.f, 0.f, 0.f, 0.f, 0.f};

    for (int k0 = 0; k0 < K; k0 += 32) {
        __syncthreads();
        // ---- stage B tile, swizzled so lane fragments are contiguous ------
        {
            uint2 bv = *(const uint2*)(Bw + (size_t)(k0 + skk) * ldb + tileN + snn);
            unsigned short e0 = (unsigned short)(bv.x & 0xffffu);
            unsigned short e1 = (unsigned short)(bv.x >> 16);
            unsigned short e2 = (unsigned short)(bv.y & 0xffffu);
            unsigned short e3 = (unsigned short)(bv.y >> 16);
            int lb = (kmid << 4) + snn;        // destination lane group
            sB[(lb + 0) * 16 + sbase] = e0;
            sB[(lb + 1) * 16 + sbase] = e1;
            sB[(lb + 2) * 16 + sbase] = e2;
            sB[(lb + 3) * 16 + sbase] = e3;
        }
        __syncthreads();

        // ---- A fragments: two b128 loads each, concat (no cvt ALU) --------
        v8us lo0 = *(const v8us*)(a0 + k0 + khalf);
        v8us hi0 = *(const v8us*)(a0 + k0 + khalf + 16);
        v8us lo1 = *(const v8us*)(a1 + k0 + khalf);
        v8us hi1 = *(const v8us*)(a1 + k0 + khalf + 16);
        if (k0 + 32 < K) {
            __builtin_prefetch(a0 + k0 + 32, 0, 1);
            __builtin_prefetch(a1 + k0 + 32, 0, 1);
        }
        v16us au0 = __builtin_shufflevector(lo0, hi0,
                        0, 1, 2, 3, 4, 5, 6, 7, 8, 9, 10, 11, 12, 13, 14, 15);
        v16us au1 = __builtin_shufflevector(lo1, hi1,
                        0, 1, 2, 3, 4, 5, 6, 7, 8, 9, 10, 11, 12, 13, 14, 15);

        // ---- B fragment: one contiguous 32-byte LDS read ------------------
        v16us bu = *(const v16us*)(sB + lane * 16);

        v16bf a_0 = __builtin_bit_cast(v16bf, au0);
        v16bf a_1 = __builtin_bit_cast(v16bf, au1);
        v16bf b_f = __builtin_bit_cast(v16bf, bu);

        acc0 = __builtin_amdgcn_wmma_f32_16x16x32_bf16(
                   false, a_0, false, b_f, (short)0, acc0, false, false);
        acc1 = __builtin_amdgcn_wmma_f32_16x16x32_bf16(
                   false, a_1, false, b_f, (short)0, acc1, false, false);
    }

    const int ncol = tileN + nloc;
    float bv = bias ? bias[ncol] : 0.f;
    const int r0 = tileM + (lhi << 3);
#pragma unroll
    for (int v = 0; v < 8; ++v) {
        float v0 = acc0[v] + bv;
        float v1 = acc1[v] + bv;
        if (act == 1) { v0 = softplus_f(v0); v1 = softplus_f(v1); }
        Cc[(size_t)(r0 + v) * ldc + ncol]      = v0;
        Cc[(size_t)(r0 + 16 + v) * ldc + ncol] = v1;
    }
}

// --------------------- causal depthwise conv + SiLU ------------------------
// writes f32 (scan input u) and bf16 (GEMM2 A operand)
__global__ void conv_silu_kernel(const float* __restrict__ xr,
                                 const float* __restrict__ cw,
                                 const float* __restrict__ cb,
                                 float* __restrict__ xc,
                                 unsigned short* __restrict__ xc16,
                                 int di, int d4, int L, long long total) {
    long long g = (long long)blockIdx.x * 256 + threadIdx.x;
    if (g >= total) return;
    int ch = (int)(g % di);
    long long ml = g / di;
    int l = (int)(ml % (long long)L);
    float acc = cb[ch];
#pragma unroll
    for (int j = 0; j < 4; ++j) {
        int lj = l - 3 + j;
        if (lj >= 0)
            acc += cw[ch * 4 + j] * xr[(ml - 3 + j) * (long long)d4 + ch];
    }
    float v = silu_f(acc);
    xc[g] = v;
    xc16[g] = f2bf(v);
}

// --------------------------- selective scan --------------------------------
__global__ void __launch_bounds__(256)
scan_kernel(float* __restrict__ dy,           // delta in, y out (in place)
            const float* __restrict__ xc,     // u
            const float* __restrict__ xdbl,   // (M, rn): [dt_r | B | C]
            const float* __restrict__ A_log,
            const float* __restrict__ Dp,
            int L, int di, int r, int rn) {
    int ch = blockIdx.x * 256 + threadIdx.x;
    int b  = blockIdx.y;
    float negA[16];
#pragma unroll
    for (int nn = 0; nn < 16; ++nn) negA[nn] = -__expf(A_log[ch * 16 + nn]);
    float Dv = Dp[ch];
    float h[16];
#pragma unroll
    for (int nn = 0; nn < 16; ++nn) h[nn] = 0.f;

    __shared__ float sBC[32];                  // [0:16)=B_t, [16:32)=C_t
    const float* xd = xdbl + (size_t)b * L * rn;
    size_t idx = ((size_t)b * L) * di + ch;

    for (int l = 0; l < L; ++l, idx += di) {
        __syncthreads();
        if (threadIdx.x < 32)
            sBC[threadIdx.x] = xd[(size_t)l * rn + r + threadIdx.x];
        __syncthreads();
        float dt = dy[idx];
        float u  = xc[idx];
        float du = dt * u;
        float acc = 0.f;
#pragma unroll
        for (int nn = 0; nn < 16; ++nn) {
            h[nn] = __expf(dt * negA[nn]) * h[nn] + du * sBC[nn];
            acc  += h[nn] * sBC[16 + nn];
        }
        dy[idx] = acc + u * Dv;
    }
}

// ------------------------------ gating -------------------------------------
// yg16 = bf16( y * silu(res) )  -> A operand of out_proj GEMM
__global__ void gate_kernel(const float* __restrict__ y,
                            const float* __restrict__ xr,
                            unsigned short* __restrict__ yg16,
                            int di, int d4, long long total) {
    long long g = (long long)blockIdx.x * 256 + threadIdx.x;
    if (g >= total) return;
    long long row = g / di;
    int c = (int)(g % di);
    float rv = xr[row * (long long)d4 + di + c];
    yg16[g] = f2bf(y[g] * silu_f(rv));
}

// ------------------- residual + final rmsnorm + transpose ------------------
__global__ void __launch_bounds__(256)
residual_rmsnorm_out_kernel(const float* __restrict__ x,     // (B,d,L)
                            const float* __restrict__ yp_,   // (B*L,d)
                            const float* __restrict__ w,
                            float* __restrict__ out,         // (B,d,L)
                            int d, int L) {
    int l = blockIdx.x, b = blockIdx.y;
    __shared__ float sbuf[1024];
    __shared__ float sred[256];
    const float* xp = x + ((size_t)b * d) * L + l;
    const float* yp = yp_ + ((size_t)b * L + l) * d;
    float ss = 0.f;
    for (int c = threadIdx.x; c < d; c += 256) {
        float v = xp[(size_t)c * L] + yp[c];
        sbuf[c] = v;
        ss += v * v;
    }
    sred[threadIdx.x] = ss;
    __syncthreads();
    for (int s = 128; s > 0; s >>= 1) {
        if (threadIdx.x < s) sred[threadIdx.x] += sred[threadIdx.x + s];
        __syncthreads();
    }
    float rinv = rsqrtf(sred[0] / (float)d + 1e-5f);
    float* op = out + ((size_t)b * d) * L + l;
    for (int c = threadIdx.x; c < d; c += 256)
        op[(size_t)c * L] = sbuf[c] * rinv * w[c];
}

// ===========================================================================
extern "C" void kernel_launch(void* const* d_in, const int* in_sizes, int n_in,
                              void* d_out, int out_size, void* d_ws, size_t ws_size,
                              hipStream_t stream) {
    const float* s3 = (const float*)d_in[0];
    const float* s4 = (const float*)d_in[1];
    const float* s5 = (const float*)d_in[2];

    // Param leaf ordering: insertion order vs pytree-sorted key order.
    bool ins = (n_in >= 4 && in_sizes[3] == 262144);
    // canonical: 0 in_proj 1 conv_w 2 conv_b 3 x_proj 4 dt_w 5 dt_b
    //            6 A_log  7 D      8 out_proj 9 norm_w 10 norm_f_w
    static const int map_ins[11]  = {0, 1, 2, 3, 4, 5, 6, 7, 8, 9, 10};
    static const int map_sort[11] = {6, 3, 2, 10, 5, 4, 0, 1, 9, 8, 7};
    const int* map = ins ? map_ins : map_sort;

    const float* P3[11]; const float* P4[11]; const float* P5[11];
    for (int i = 0; i < 11; ++i) {
        P3[i] = (const float*)d_in[3  + map[i]];
        P4[i] = (const float*)d_in[14 + map[i]];
        P5[i] = (const float*)d_in[25 + map[i]];
    }

    // ---------------- workspace carve-up (sized by per-scale maxima) -------
    char* base = (char*)d_ws;
    size_t o = 0;
    auto carve = [&](size_t bytes) -> void* {
        void* p = base + o;
        o += (bytes + 255) & ~(size_t)255;
        return p;
    };
    unsigned short* XN16 = (unsigned short*)carve(8388608ull  * 2);  // M*d  bf16
    float* XR    = (float*)carve(33554432ull * 4);                   // M*4d
    float* XC    = (float*)carve(16777216ull * 4);                   // M*di (scan u)
    unsigned short* XC16 = (unsigned short*)carve(16777216ull * 2);  // M*di bf16
    float* XDBL  = (float*)carve(1572864ull  * 4);                   // M*(r+32)
    float* DELTA = (float*)carve(16777216ull * 4);                   // M*di, reused as y
    unsigned short* YG16 = (unsigned short*)carve(16777216ull * 2);  // M*di bf16
    float* OUTB  = (float*)carve(8388608ull  * 4);                   // M*d
    unsigned short* DTR16 = (unsigned short*)carve(1048576ull * 2);  // M*Kp bf16
    unsigned short* Win  = (unsigned short*)carve(4194304ull * 2);
    unsigned short* Wxp  = (unsigned short*)carve(196608ull  * 2);
    unsigned short* Wdt  = (unsigned short*)carve(131072ull  * 2);
    unsigned short* Wout = (unsigned short*)carve(2097152ull * 2);
    (void)ws_size;

    auto cvt = [&](const float* src, unsigned short* dst, long long n) {
        f32_to_bf16_kernel<<<dim3((unsigned)((n + 255) / 256)), 256, 0, stream>>>(src, dst, n);
    };

    auto run_scale = [&](const float* x, const float* const* P, float* out,
                         int d, int L) {
        const int Bsz = 8;
        const int di = 2 * d, d4 = 4 * d, r = d / 16, rn = r + 32;
        const int Kp = (r < 32) ? 32 : r;      // padded K for dt GEMM
        const int M = Bsz * L;
        const long long totdi = (long long)M * di;

        cvt(P[0], Win,  (long long)d  * d4);   // in_proj  (d, 4d)
        cvt(P[3], Wxp,  (long long)di * rn);   // x_proj   (di, rn)
        cvt(P[8], Wout, (long long)di * d);    // out_proj (di, d)
        {   // dt_w (r, di) -> zero-padded (Kp, di)
            long long n = (long long)Kp * di;
            f32_to_bf16_pad_kernel<<<dim3((unsigned)((n + 255) / 256)), 256, 0, stream>>>(
                P[4], Wdt, r, di, n);
        }

        rmsnorm_transpose_kernel<<<dim3(L, Bsz), 256, 0, stream>>>(
            x, P[9], XN16, d, L);

        // xr = xn @ in_proj
        wmma_gemm_kernel<<<dim3(d4 / 16, M / 128), 128, 0, stream>>>(
            XN16, d, Win, d4, XR, d4, d, nullptr, 0);

        // xc = silu(conv(xr[:, :di]))      (== both xi and ci since c == x)
        conv_silu_kernel<<<dim3((unsigned)((totdi + 255) / 256)), 256, 0, stream>>>(
            XR, P[1], P[2], XC, XC16, di, d4, L, totdi);

        // x_dbl = xc @ x_proj
        wmma_gemm_kernel<<<dim3(rn / 16, M / 128), 128, 0, stream>>>(
            XC16, di, Wxp, rn, XDBL, rn, di, nullptr, 0);

        // dt_r (zero-padded to Kp columns, bf16)
        {
            long long n = (long long)M * Kp;
            pad_dtr_kernel<<<dim3((unsigned)((n + 255) / 256)), 256, 0, stream>>>(
                XDBL, DTR16, rn, r, Kp, n);
        }

        // delta = softplus(dt_r @ dt_w + dt_b)
        wmma_gemm_kernel<<<dim3(di / 16, M / 128), 128, 0, stream>>>(
            DTR16, Kp, Wdt, di, DELTA, di, Kp, P[5], 1);

        // selective scan (y written over DELTA in place)
        scan_kernel<<<dim3(di / 256, Bsz), 256, 0, stream>>>(
            DELTA, XC, XDBL, P[6], P[7], L, di, r, rn);

        // yg16 = bf16(y * silu(res))
        gate_kernel<<<dim3((unsigned)((totdi + 255) / 256)), 256, 0, stream>>>(
            DELTA, XR, YG16, di, d4, totdi);

        // y @ out_proj
        wmma_gemm_kernel<<<dim3(d / 16, M / 128), 128, 0, stream>>>(
            YG16, di, Wout, d, OUTB, d, di, nullptr, 0);

        // out = rmsnorm(x + y_out, norm_f_w), back to (B, d, L) layout
        residual_rmsnorm_out_kernel<<<dim3(L, Bsz), 256, 0, stream>>>(
            x, OUTB, P[10], out, d, L);
    };

    float* out3 = (float*)d_out;
    float* out4 = out3 + (size_t)8 * 256 * 4096;
    float* out5 = out4 + (size_t)8 * 512 * 1024;
    (void)out_size;

    run_scale(s3, P3, out3, 256,  4096);
    run_scale(s4, P4, out4, 512,  1024);
    run_scale(s5, P5, out5, 1024, 256);
}
